// SglangFusedMoE_65352222375984
// MI455X (gfx1250) — matrix operand order repaired and checked by reading, not compiled, not run on hardware
//
#include <hip/hip_runtime.h>

#define TOKENS 4096
#define HIDDEN 2048
#define INTER  4096
#define NEXP   8
#define ROWS   (TOKENS * 2)   // top_k = 2 -> every token occupies exactly 2 rows

typedef __attribute__((ext_vector_type(16))) __bf16 v16bf;
typedef __attribute__((ext_vector_type(8)))  float  v8f;
typedef __attribute__((ext_vector_type(2)))  float  f32x2;   // trivial type: AS-cast friendly

// ---------- helpers ----------

#if defined(__HIP_DEVICE_COMPILE__) && __has_builtin(__builtin_amdgcn_cvt_pk_bf16_f32)
__device__ __forceinline__ unsigned int pack_bf16x2(float a, float b) {
    auto pv = __builtin_amdgcn_cvt_pk_bf16_f32(a, b);   // v_cvt_pk_bf16_f32
    unsigned int u;
    __builtin_memcpy(&u, &pv, 4);
    return u;
}
#else
__device__ __forceinline__ unsigned int pack_bf16x2(float a, float b) {
    unsigned int ua = __float_as_uint(a), ub = __float_as_uint(b);
    ua += 0x7FFFu + ((ua >> 16) & 1u);          // round-to-nearest-even
    ub += 0x7FFFu + ((ub >> 16) & 1u);
    return (ua >> 16) | (ub & 0xFFFF0000u);
}
#endif

__device__ __forceinline__ unsigned short f2bf(float a) {
    unsigned int ua = __float_as_uint(a);
    ua += 0x7FFFu + ((ua >> 16) & 1u);
    return (unsigned short)(ua >> 16);
}

// Explicit global (addrspace 1) loads: guarantees global_load_* encoding so the
// gathered A rows never fall back to flat_load (FLAT pollutes DScnt and forces
// full s_wait_dscnt before every WMMA).  Uses an ext_vector (primitive) type so
// the AS-qualified dereference needs no constructor.
__device__ __forceinline__ f32x2 gload_f2(unsigned long long addr, unsigned long long off) {
    return *reinterpret_cast<const f32x2 __attribute__((address_space(1)))*>(addr + off);
}

// Load a 16-element bf16 WMMA fragment (8 VGPRs) from LDS as two b128 reads.
__device__ __forceinline__ v16bf ld_frag(const unsigned short* smem, int off0, int off1) {
    union { v16bf v; uint4 u[2]; } f;
    f.u[0] = *reinterpret_cast<const uint4*>(reinterpret_cast<const char*>(smem) + off0);
    f.u[1] = *reinterpret_cast<const uint4*>(reinterpret_cast<const char*>(smem) + off1);
    return f.v;
}

// Async global->LDS b128 (ASYNCcnt-tracked).  VDST = LDS byte offset (addr[31:0] of
// the generic shared pointer), VADDR = 64-bit global address, saddr = off.
__device__ __forceinline__ void async_copy_b128(unsigned lds_byte_off, unsigned long long gsrc) {
#if defined(__HIP_DEVICE_COMPILE__)
    asm volatile("global_load_async_to_lds_b128 %0, %1, off"
                 :: "v"(lds_byte_off), "v"(gsrc)
                 : "memory");
#endif
}
__device__ __forceinline__ void wait_async0() {
#if defined(__HIP_DEVICE_COMPILE__)
    asm volatile("s_wait_asynccnt 0x0" ::: "memory");
#endif
}

__device__ __forceinline__ v8f wmma_bf16(v16bf a, v16bf b, v8f c) {
    return __builtin_amdgcn_wmma_f32_16x16x32_bf16(false, a, false, b, (short)0, c, false, false);
}

// ---------- routing ----------

__global__ void moe_init(int* counts) {
    if (threadIdx.x < NEXP) counts[threadIdx.x] = 0;
}

__global__ __launch_bounds__(256)
void moe_router(const float* __restrict__ logits, int2* __restrict__ eid,
                float2* __restrict__ gw, int* __restrict__ counts) {
    int t = blockIdx.x * 256 + threadIdx.x;
    const float* l = logits + (size_t)t * NEXP;
    float v0 = l[0], v1 = -3.4e38f;
    int b0 = 0, b1 = 0;
#pragma unroll
    for (int e = 1; e < NEXP; ++e) {
        float x = l[e];
        if (x > v0)      { v1 = v0; b1 = b0; v0 = x; b0 = e; }
        else if (x > v1) { v1 = x;  b1 = e; }
    }
    // normalized top-2 softmax weights in closed form
    float tt  = __expf(v1 - v0);
    float inv = __builtin_amdgcn_rcpf(1.f + tt);
    eid[t] = make_int2(b0, b1);
    gw[t]  = make_float2(inv, tt * inv);
    atomicAdd(&counts[b0], 1);     // integer atomics: order-independent result
    atomicAdd(&counts[b1], 1);
}

// Deterministic counting-sort scatter: one block, stable order by token index.
__global__ __launch_bounds__(256)
void moe_scatter(const int2* __restrict__ eid, const float2* __restrict__ gw,
                 const int* __restrict__ counts, int* __restrict__ offs,
                 int* __restrict__ tok_of_row, float* __restrict__ gate_of_row,
                 int2* __restrict__ slot2row) {
    __shared__ int soffs[NEXP];
    __shared__ int run[NEXP];
    __shared__ int sc[256];
    const int tid = threadIdx.x;
    if (tid == 0) {
        int acc = 0;
        for (int e = 0; e < NEXP; ++e) { soffs[e] = acc; offs[e] = acc; run[e] = 0; acc += counts[e]; }
        offs[NEXP] = acc;
    }
    __syncthreads();
    for (int c = 0; c < TOKENS / 256; ++c) {
        int t = c * 256 + tid;
        int2   ee = eid[t];
        float2 ww = gw[t];
        for (int e = 0; e < NEXP; ++e) {
            int slot = (ee.x == e) ? 0 : ((ee.y == e) ? 1 : -1);
            int f = (slot >= 0) ? 1 : 0;
            sc[tid] = f; __syncthreads();
            for (int s = 1; s < 256; s <<= 1) {              // Hillis-Steele inclusive scan
                int mine = sc[tid];
                int add  = (tid >= s) ? sc[tid - s] : 0;
                __syncthreads();
                sc[tid] = mine + add;
                __syncthreads();
            }
            int excl = sc[tid] - f;
            int tot  = sc[255];
            if (f) {
                int row = soffs[e] + run[e] + excl;
                tok_of_row[row]  = t;
                gate_of_row[row] = (slot == 0) ? ww.x : ww.y;
                if (slot == 0) slot2row[t].x = row; else slot2row[t].y = row;
            }
            __syncthreads();
            if (tid == 0) run[e] += tot;
            __syncthreads();
        }
    }
}

// ---------- GEMM1: act[row, i] = gate * silu(x·w13g) * (x·w13u), bf16 out ----------
// Block tile: M=64 rows x N=128 inter cols (gate AND up). 8 waves: 4(M) x 2(N).
// Double-buffered LDS, ONE barrier per K-step.  waves_per_eu(1,4) widens the VGPR
// budget so all 8 B fragments stay live and ds_loads retire under staggered waits.

#define BUF1 ((64 + 256) * 40)          // halfs per buffer (pitch 40 halfs = 80B)

__global__ __launch_bounds__(256)
__attribute__((amdgpu_waves_per_eu(1, 4)))
void moe_gemm1(const float* __restrict__ hs, const float* __restrict__ w13,
               const int* __restrict__ tok_of_row, const float* __restrict__ gate_of_row,
               const int* __restrict__ counts, const int* __restrict__ offs,
               unsigned short* __restrict__ act) {
    const int e      = blockIdx.z;
    const int cnt    = counts[e];
    const int m_tile = blockIdx.y * 64;
    if (m_tile >= cnt) return;
    const int n_base   = blockIdx.x * 128;
    const int row_base = offs[e];

    __shared__ __align__(16) unsigned short smem[2 * BUF1];
    __shared__ int stok[64];

    const int tid = threadIdx.x;
    if (tid < 64) {
        int r = m_tile + tid;
        stok[tid] = tok_of_row[row_base + ((r < cnt) ? r : 0)];
    }
    __syncthreads();

    const int wv = tid >> 5, lane = tid & 31;
    const int wm = wv & 3, wn = wv >> 2;
    const int sel = lane >> 4, nl = lane & 15;
    const int t4 = tid >> 4;          // staging row-group 0..15
    const int k2 = (tid & 15) * 2;    // staging K offset 0..30

    // Integer global addresses (loads forced to addrspace(1))
    unsigned long long aadr[4];
#pragma unroll
    for (int j = 0; j < 4; ++j)
        aadr[j] = (unsigned long long)(uintptr_t)(hs + (size_t)stok[j * 16 + t4] * HIDDEN + k2);
    const unsigned long long bgadr =
        (unsigned long long)(uintptr_t)(w13 + ((size_t)e * (2 * INTER) + n_base + t4) * HIDDEN + k2);
    const unsigned long long buadr = bgadr + (unsigned long long)INTER * HIDDEN * 4;

    // Stage one 32-wide K slab into buffer b (rows at immediate offsets j*16*K*4B).
    auto stage = [&](int k0, int b) {
        unsigned short* sA = smem + b * BUF1;
        unsigned short* sB = sA + 64 * 40;
        const unsigned long long kb = (unsigned long long)k0 * 4;
#pragma unroll
        for (int j = 0; j < 4; ++j) {
            const f32x2 d = gload_f2(aadr[j], kb);
            *reinterpret_cast<unsigned int*>(&sA[(j * 16 + t4) * 40 + k2]) = pack_bf16x2(d.x, d.y);
        }
#pragma unroll
        for (int j = 0; j < 8; ++j) {
            const f32x2 dg = gload_f2(bgadr, kb + (unsigned long long)j * 16 * HIDDEN * 4);
            const f32x2 du = gload_f2(buadr, kb + (unsigned long long)j * 16 * HIDDEN * 4);
            *reinterpret_cast<unsigned int*>(&sB[(j * 16 + t4) * 40 + k2])       = pack_bf16x2(dg.x, dg.y);
            *reinterpret_cast<unsigned int*>(&sB[(128 + j * 16 + t4) * 40 + k2]) = pack_bf16x2(du.x, du.y);
        }
    };

    v8f accg[4] = {};
    v8f accu[4] = {};

    const int a0    = (wm * 16 + nl) * 80 + sel * 32;
    const int brow0 = wn * 64 + nl;

    stage(0, 0);
    int cur = 0;
#pragma unroll 2
    for (int k0 = 0; k0 < HIDDEN; k0 += 32, cur ^= 1) {
        __syncthreads();                       // publish tile k0; protect tile k0-32's buffer
        if (k0 + 32 < HIDDEN) stage(k0 + 32, cur ^ 1);

        const unsigned short* sb = smem + cur * BUF1;
        v16bf afrag = ld_frag(sb, a0, a0 + 16);
        v16bf bf[8];
#pragma unroll
        for (int i = 0; i < 4; ++i) {
            int g = (64 + brow0 + i * 16) * 80 + sel * 32;
            bf[i]     = ld_frag(sb, g, g + 16);
            bf[4 + i] = ld_frag(sb, g + 128 * 80, g + 128 * 80 + 16);
        }
#pragma unroll
        for (int i = 0; i < 4; ++i) {
            accg[i] = wmma_bf16(afrag, bf[i],     accg[i]);
            accu[i] = wmma_bf16(afrag, bf[4 + i], accu[i]);
        }
    }

    // epilogue: silu(g)*u * gate -> bf16 act (gate folded here so GEMM2 output just sums)
#pragma unroll
    for (int v = 0; v < 8; ++v) {
        int mrow = m_tile + wm * 16 + sel * 8 + v;   // C layout: M = v + 8*(lane>=16)
        if (mrow < cnt) {
            float gwv = gate_of_row[row_base + mrow];
            size_t rowoff = (size_t)(row_base + mrow) * INTER;
#pragma unroll
            for (int i = 0; i < 4; ++i) {
                int col = n_base + wn * 64 + i * 16 + nl;
                float g = accg[i][v], u = accu[i][v];
                float s = g * __builtin_amdgcn_rcpf(1.f + __expf(-g));   // fast silu
                act[rowoff + col] = f2bf(s * u * gwv);
            }
        }
    }
}

// ---------- GEMM2: ybuf[row, h] = act[row,:] · w2[e][h,:] ----------
// A tile (already bf16 in memory) staged with GLOBAL_LOAD_ASYNC_TO_LDS_B128.
// Double-buffered LDS, one barrier per K-step; s_wait_asynccnt 0 before the barrier.

#define BUF2 ((64 + 128) * 40)

__global__ __launch_bounds__(256)
__attribute__((amdgpu_waves_per_eu(1, 4)))
void moe_gemm2(const unsigned short* __restrict__ act, const float* __restrict__ w2,
               const int* __restrict__ counts, const int* __restrict__ offs,
               float* __restrict__ ybuf) {
    const int e      = blockIdx.z;
    const int cnt    = counts[e];
    const int m_tile = blockIdx.y * 64;
    if (m_tile >= cnt) return;
    const int n_base   = blockIdx.x * 128;
    const int row_base = offs[e];

    __shared__ __align__(16) unsigned short smem[2 * BUF2];
    const int tid = threadIdx.x;
    const int wv = tid >> 5, lane = tid & 31;
    const int wm = wv & 3, wn = wv >> 2;
    const int sel = lane >> 4, nl = lane & 15;
    const int t4 = tid >> 4;
    const int k2 = (tid & 15) * 2;

    // Per-thread async copy assignment for the A tile: 64 rows x 64B = 256 b128 segments.
    const int ar  = tid >> 2;                 // row 0..63
    const int seg = tid & 3;                  // 16B segment 0..3
    int arr = m_tile + ar; if (arr >= cnt) arr = 0;
    const unsigned long long aadr = (unsigned long long)(uintptr_t)(
        act + (size_t)(row_base + arr) * INTER) + seg * 16;
    const unsigned lds_a0 = (unsigned)(uintptr_t)&smem[ar * 40] + seg * 16;

    const unsigned long long badr =
        (unsigned long long)(uintptr_t)(w2 + ((size_t)e * HIDDEN + n_base + t4) * INTER + k2);

    auto stage = [&](int k0, int b) {
        // A: bf16 rows, async global->LDS (no VGPR round trip, ASYNCcnt-tracked)
        async_copy_b128(lds_a0 + b * (BUF2 * 2), aadr + (unsigned long long)k0 * 2);
        // B: 128 rows of w2[e], f32 -> bf16
        unsigned short* sB = smem + b * BUF2 + 64 * 40;
        const unsigned long long kb = (unsigned long long)k0 * 4;
#pragma unroll
        for (int j = 0; j < 8; ++j) {
            const f32x2 d = gload_f2(badr, kb + (unsigned long long)j * 16 * INTER * 4);
            *reinterpret_cast<unsigned int*>(&sB[(j * 16 + t4) * 40 + k2]) = pack_bf16x2(d.x, d.y);
        }
    };

    v8f acc[4] = {};
    const int a0    = (wm * 16 + nl) * 80 + sel * 32;
    const int brow0 = wn * 64 + nl;

    stage(0, 0);
    int cur = 0;
#pragma unroll 2
    for (int k0 = 0; k0 < INTER; k0 += 32, cur ^= 1) {
        wait_async0();                         // this wave's async segments are in LDS
        __syncthreads();                       // everyone's segments visible
        if (k0 + 32 < INTER) stage(k0 + 32, cur ^ 1);

        const unsigned short* sb = smem + cur * BUF2;
        v16bf afrag = ld_frag(sb, a0, a0 + 16);
        v16bf bf[4];
#pragma unroll
        for (int i = 0; i < 4; ++i) {
            int g = (64 + brow0 + i * 16) * 80 + sel * 32;
            bf[i] = ld_frag(sb, g, g + 16);
        }
#pragma unroll
        for (int i = 0; i < 4; ++i)
            acc[i] = wmma_bf16(afrag, bf[i], acc[i]);
    }

#pragma unroll
    for (int v = 0; v < 8; ++v) {
        int mrow = m_tile + wm * 16 + sel * 8 + v;
        if (mrow < cnt) {
            size_t rowoff = (size_t)(row_base + mrow) * HIDDEN;
#pragma unroll
            for (int i = 0; i < 4; ++i) {
                int col = n_base + wn * 64 + i * 16 + nl;
                ybuf[rowoff + col] = acc[i][v];
            }
        }
    }
}

// ---------- combine: out[t] = y[row0] + y[row1] (fixed order -> deterministic) ----------

__global__ __launch_bounds__(256)
void moe_combine(const float* __restrict__ ybuf, const int2* __restrict__ slot2row,
                 float* __restrict__ out) {
    int i = blockIdx.x * 256 + threadIdx.x;        // float4 index
    int t = i / (HIDDEN / 4);
    int c = i % (HIDDEN / 4);
    int2 rr = slot2row[t];
    float4 a = reinterpret_cast<const float4*>(ybuf)[(size_t)rr.x * (HIDDEN / 4) + c];
    float4 b = reinterpret_cast<const float4*>(ybuf)[(size_t)rr.y * (HIDDEN / 4) + c];
    float4 r; r.x = a.x + b.x; r.y = a.y + b.y; r.z = a.z + b.z; r.w = a.w + b.w;
    reinterpret_cast<float4*>(out)[i] = r;
}

// ---------- launch ----------

extern "C" void kernel_launch(void* const* d_in, const int* in_sizes, int n_in,
                              void* d_out, int out_size, void* d_ws, size_t ws_size,
                              hipStream_t stream) {
    const float* hs     = (const float*)d_in[0];
    const float* logits = (const float*)d_in[1];
    const float* w13    = (const float*)d_in[2];
    const float* w2     = (const float*)d_in[3];
    // d_in[4] = top_k (fixed at 2 in this problem size)
    float* out = (float*)d_out;

    char* w = (char*)d_ws;
    int*    counts      = (int*)w;            w += 256;
    int*    offs        = (int*)w;            w += 256;
    int2*   eid         = (int2*)w;           w += (size_t)TOKENS * sizeof(int2);
    float2* gwts        = (float2*)w;         w += (size_t)TOKENS * sizeof(float2);
    int*    tok_of_row  = (int*)w;            w += (size_t)ROWS * sizeof(int);
    float*  gate_of_row = (float*)w;          w += (size_t)ROWS * sizeof(float);
    int2*   slot2row    = (int2*)w;           w += (size_t)TOKENS * sizeof(int2);
    unsigned short* act = (unsigned short*)w; w += (size_t)ROWS * INTER * sizeof(unsigned short);
    float*  ybuf        = (float*)w;          w += (size_t)ROWS * HIDDEN * sizeof(float);
    (void)ws_size; (void)in_sizes; (void)n_in; (void)out_size;

    moe_init   <<<1, 32, 0, stream>>>(counts);
    moe_router <<<TOKENS / 256, 256, 0, stream>>>(logits, eid, gwts, counts);
    moe_scatter<<<1, 256, 0, stream>>>(eid, gwts, counts, offs, tok_of_row, gate_of_row, slot2row);
    moe_gemm1  <<<dim3(INTER / 128, TOKENS / 64, NEXP), 256, 0, stream>>>(
        hs, w13, tok_of_row, gate_of_row, counts, offs, act);
    moe_gemm2  <<<dim3(HIDDEN / 128, TOKENS / 64, NEXP), 256, 0, stream>>>(
        act, w2, counts, offs, ybuf);
    moe_combine<<<(TOKENS * HIDDEN / 4) / 256, 256, 0, stream>>>(ybuf, slot2row, out);
}
